// CausalSelfAttention_10599979286823
// MI455X (gfx1250) — compile-verified, matrix operand order checked by reading, not run on hardware
//
#include <hip/hip_runtime.h>
#include <hip/hip_bf16.h>

// ---------------------------------------------------------------------------
// Causal self-attention for MI455X (gfx1250): bf16 WMMA everywhere,
// flash-attention streaming softmax, TDM (tensor_load_to_lds) staged GEMMs.
// ---------------------------------------------------------------------------

#define D_MODEL 1024
#define N_HEADS 16
#define HEAD_DIM 64
#define SEQ_T 2048
#define BATCH 4
#define ROWS (BATCH * SEQ_T)          // 8192
#define QKV_N (3 * D_MODEL)           // 3072

typedef __attribute__((ext_vector_type(16))) __bf16 v16bf;
typedef __attribute__((ext_vector_type(8)))  float  v8f;
typedef __attribute__((ext_vector_type(4)))  unsigned int u32x4;
typedef __attribute__((ext_vector_type(8)))  int   i32x8;
typedef __attribute__((ext_vector_type(4)))  int   i32x4;

__device__ __forceinline__ unsigned short f2bf(float x) {
    unsigned u = __float_as_uint(x);
    unsigned r = u + 0x7FFFu + ((u >> 16) & 1u);   // round-to-nearest-even
    return (unsigned short)(r >> 16);
}

// Load a 16x32 bf16 fragment in the CDNA5 16-bit A/B WMMA layout.
// lane L: row = L%16 ; K-half = (L<16 ? 0 : 8); elements 0..7 -> K=kh+0..7,
// elements 8..15 -> K=kh+16..23. Works for global or LDS-backed pointers
// (two contiguous 16-byte loads per lane -> b128 / ds_load_b128).
__device__ __forceinline__ v16bf load_frag(const unsigned short* __restrict__ base,
                                           int row0, int lda, int k0) {
    int lane = threadIdx.x & 31;
    int r    = lane & 15;
    int kh   = (lane >> 4) << 3;   // 0 or 8
    const unsigned short* p = base + (size_t)(row0 + r) * lda + (k0 + kh);
    union { u32x4 u[2]; v16bf v; } f;
    f.u[0] = *(const u32x4*)(p);
    f.u[1] = *(const u32x4*)(p + 16);
    return f.v;
}

__device__ __forceinline__ v8f bwmma(v16bf a, v16bf b, v8f c) {
    return __builtin_amdgcn_wmma_f32_16x16x32_bf16(
        /*neg_a=*/false, a, /*neg_b=*/false, b,
        /*c_mod=*/(short)0, c, /*reuse_a=*/false, /*reuse_b=*/false);
}

// ---------------------------------------------------------------------------
// Tensor Data Mover: async DMA of a 2-D bf16 tile [tile_rows x tile_cols]
// from a row-major [rows_total x lda] tensor into LDS (contiguous tile).
// Builds the D# per CDNA5 ISA 8.3/8.4: group0 = {flags, lds_addr,
// global_addr, type=2}; group1 = {data_size=2B, dims, tile dims, strides}.
// Tracked by TENSORcnt (wave-level, EXEC-independent).
// This toolchain exposes the 6-arg builtin:
//   (uint32x4 g0, int32x8 g1, int32x4 g2, int32x4 g3, int32x8 g4, i32 cpol)
// ---------------------------------------------------------------------------
__device__ __forceinline__ void tdm_load_tile_2d(const unsigned short* gptr,
                                                 unsigned lds_byte_off,
                                                 int tile_rows, int tile_cols,
                                                 int lda, int rows_total) {
    unsigned long long ga = (unsigned long long)(uintptr_t)gptr;
    u32x4 g0;
    g0[0] = 1u;                                              // count=1, user mode
    g0[1] = lds_byte_off;                                    // LDS dest (bytes)
    g0[2] = (unsigned)(ga & 0xFFFFFFFFu);                    // global_addr[31:0]
    g0[3] = (unsigned)((ga >> 32) & 0x01FFFFFFu) | (2u << 30); // addr[56:32], type=2
    i32x8 g1;
    g1[0] = (1 << 16);                                       // data_size=1 -> 2 bytes
    g1[1] = (int)(((unsigned)lda & 0xFFFFu) << 16);          // tensor_dim0[15:0]
    g1[2] = (int)((((unsigned)lda >> 16) & 0xFFFFu)          // tensor_dim0[31:16]
                  | (((unsigned)rows_total & 0xFFFFu) << 16)); // tensor_dim1[15:0]
    g1[3] = (int)((((unsigned)rows_total >> 16) & 0xFFFFu)   // tensor_dim1[31:16]
                  | ((unsigned)tile_cols << 16));            // tile_dim0
    g1[4] = tile_rows;                                       // tile_dim1 (tile_dim2=0)
    g1[5] = lda;                                             // tensor_dim0_stride[31:0]
    g1[6] = 0;
    g1[7] = 0;
    i32x4 z4 = {0, 0, 0, 0};
    i32x8 z8 = {0, 0, 0, 0, 0, 0, 0, 0};
    __builtin_amdgcn_tensor_load_to_lds(g0, g1, z4, z4, z8, 0);
}

// ---------------------------------------------------------------------------
// Prep kernels: f32 -> bf16 convert, and transpose-convert for weights
// ---------------------------------------------------------------------------
__global__ void f32_to_bf16_kernel(const float* __restrict__ src,
                                   unsigned short* __restrict__ dst, int n) {
    int i = blockIdx.x * blockDim.x + threadIdx.x;
    if (i < n) dst[i] = f2bf(src[i]);
}

// src is [R, C] row-major f32; dst is [C, R] row-major bf16 (B stored N-major)
__global__ void transpose_to_bf16_kernel(const float* __restrict__ src,
                                         unsigned short* __restrict__ dst,
                                         int R, int C) {
    int i = blockIdx.x * blockDim.x + threadIdx.x;
    if (i < R * C) {
        int r = i / C;
        int c = i - r * C;
        dst[(size_t)c * R + r] = f2bf(src[i]);
    }
}

// ---------------------------------------------------------------------------
// QKV GEMM: [8192,1024]bf16 x [1024,3072] (stored [3072,1024] bf16) + bias.
// Block = 8 waves sharing one 32-row A tile (TDM double-buffered in LDS);
// each wave owns a 32x64 output tile (8 accumulators, 8 WMMA per K=32 step).
// Epilogue scatters into Q [BH,T,64], K [BH,T,64], V-transposed [BH,64,T].
// ---------------------------------------------------------------------------
__global__ __launch_bounds__(256) void qkv_gemm_kernel(
    const unsigned short* __restrict__ xb,     // [8192,1024]
    const unsigned short* __restrict__ wT,     // [3072,1024]
    const float* __restrict__ bias,            // [3072]
    unsigned short* __restrict__ qbuf,
    unsigned short* __restrict__ kbuf,
    unsigned short* __restrict__ vTbuf) {
    const int K = D_MODEL;
    __shared__ unsigned short abuf[2][32][64];   // 2 x 4KB double buffer

    int wave = threadIdx.x >> 5;
    int lane = threadIdx.x & 31;
    int m0 = blockIdx.x * 32;
    int n0 = blockIdx.y * 512 + wave * 64;

    v8f acc[2][4] = {};

    // prologue: TDM fetch of A chunk 0 (rows m0..m0+31, k 0..63)
    if (wave == 0)
        tdm_load_tile_2d(xb + (size_t)m0 * K,
                         (unsigned)(uintptr_t)(void*)&abuf[0][0][0],
                         32, 64, K, ROWS);

    for (int kc = 0; kc < K; kc += 64) {
        int buf = (kc >> 6) & 1;
        if (wave == 0) __builtin_amdgcn_s_wait_tensorcnt(0);
        __syncthreads();   // TDM data visible to all waves; prior reads retired
        if (wave == 0 && (kc + 64) < K)
            tdm_load_tile_2d(xb + (size_t)m0 * K + kc + 64,
                             (unsigned)(uintptr_t)(void*)&abuf[buf ^ 1][0][0],
                             32, 64, K, ROWS);
        if ((kc + 64) < K)   // prefetch next B chunk into caches
            __builtin_prefetch(wT + (size_t)(n0 + (lane & 63)) * K + kc + 64, 0, 0);

        const unsigned short* A = &abuf[buf][0][0];
#pragma unroll
        for (int ks = 0; ks < 64; ks += 32) {
            v16bf a0 = load_frag(A, 0,  64, ks);
            v16bf a1 = load_frag(A, 16, 64, ks);
#pragma unroll
            for (int nt = 0; nt < 4; ++nt) {
                v16bf b = load_frag(wT, n0 + nt * 16, K, kc + ks);
                acc[0][nt] = bwmma(a0, b, acc[0][nt]);
                acc[1][nt] = bwmma(a1, b, acc[1][nt]);
            }
        }
    }

    int coln   = lane & 15;
    int rowoff = (lane >> 4) << 3;   // 0 or 8
#pragma unroll
    for (int mt = 0; mt < 2; ++mt) {
#pragma unroll
        for (int nt = 0; nt < 4; ++nt) {
            int ccol = n0 + nt * 16 + coln;
            float bv = bias[ccol];
            int s   = ccol >> 10;          // 0=Q 1=K 2=V
            int rem = ccol & 1023;
            int h   = rem >> 6;
            int d   = rem & 63;
#pragma unroll
            for (int e = 0; e < 8; ++e) {
                int row = m0 + mt * 16 + rowoff + e;
                int b   = row >> 11;       // row / 2048
                int t   = row & 2047;
                unsigned short bf = f2bf(acc[mt][nt][e] + bv);
                size_t bh = (size_t)(b * N_HEADS + h);
                if (s == 0)      qbuf[(bh * SEQ_T + t) * HEAD_DIM + d] = bf;
                else if (s == 1) kbuf[(bh * SEQ_T + t) * HEAD_DIM + d] = bf;
                else             vTbuf[(bh * HEAD_DIM + d) * SEQ_T + t] = bf;
            }
        }
    }
}

// ---------------------------------------------------------------------------
// Flash attention: one 16-row q-tile per wave, streaming 32-wide k-chunks.
// QK^T -> online softmax (shfl reductions over 16-lane halves) -> LDS
// C-layout->A-layout transpose -> PV. Causal: only iterate k <= q.
// ---------------------------------------------------------------------------
__global__ __launch_bounds__(256) void flash_attn_kernel(
    const unsigned short* __restrict__ qbuf,   // [BH, T, 64]
    const unsigned short* __restrict__ kbuf,   // [BH, T, 64]
    const unsigned short* __restrict__ vT,     // [BH, 64, T]
    unsigned short* __restrict__ ybuf) {       // [B, T, H, 64] bf16
    __shared__ float lds[8][16 * 32];          // 2KB transpose tile per wave

    int wave = threadIdx.x >> 5;
    int lane = threadIdx.x & 31;
    int bh = blockIdx.y;
    int b  = bh >> 4;
    int h  = bh & 15;
    int q0 = blockIdx.x * 128 + wave * 16;

    const unsigned short* Q  = qbuf + (size_t)bh * SEQ_T * HEAD_DIM;
    const unsigned short* Kp = kbuf + (size_t)bh * SEQ_T * HEAD_DIM;
    const unsigned short* Vp = vT   + (size_t)bh * HEAD_DIM * SEQ_T;
    float* pl = &lds[wave][0];

    int m  = lane & 15;
    int kh = (lane >> 4) << 3;     // 0 or 8

    v16bf qa0 = load_frag(Q, q0, HEAD_DIM, 0);
    v16bf qa1 = load_frag(Q, q0, HEAD_DIM, 32);

    v8f o0 = {}, o1 = {}, o2 = {}, o3 = {};
    float mrow[8], lrow[8];
#pragma unroll
    for (int e = 0; e < 8; ++e) { mrow[e] = -1e30f; lrow[e] = 0.f; }

    int kend = q0 + 16;            // exclusive causal limit
    for (int k0 = 0; k0 < kend; k0 += 32) {
        // ---- scores: 16x32 tile as two 16x16 WMMA outputs
        v8f s0 = {}, s1 = {};
        {
            v16bf kb = load_frag(Kp, k0, HEAD_DIM, 0);
            s0 = bwmma(qa0, kb, s0);
            kb = load_frag(Kp, k0, HEAD_DIM, 32);
            s0 = bwmma(qa1, kb, s0);
            kb = load_frag(Kp, k0 + 16, HEAD_DIM, 0);
            s1 = bwmma(qa0, kb, s1);
            kb = load_frag(Kp, k0 + 16, HEAD_DIM, 32);
            s1 = bwmma(qa1, kb, s1);
        }

        // ---- scale + causal mask + per-chunk row max (C layout:
        //      lane holds column n=m, rows kh+e)
        float rmax[8];
#pragma unroll
        for (int e = 0; e < 8; ++e) {
            int qrow = q0 + kh + e;
            float v0 = s0[e] * 0.125f;           // 1/sqrt(64)
            float v1 = s1[e] * 0.125f;
            if (k0 + m      > qrow) v0 = -1e30f;
            if (k0 + 16 + m > qrow) v1 = -1e30f;
            s0[e] = v0; s1[e] = v1;
            float v = fmaxf(v0, v1);
            v = fmaxf(v, __shfl_xor(v, 1, 32));
            v = fmaxf(v, __shfl_xor(v, 2, 32));
            v = fmaxf(v, __shfl_xor(v, 4, 32));
            v = fmaxf(v, __shfl_xor(v, 8, 32));
            rmax[e] = v;
        }

        // ---- online softmax update
        float alpha[8];
#pragma unroll
        for (int e = 0; e < 8; ++e) {
            float mnew = fmaxf(mrow[e], rmax[e]);
            alpha[e] = __expf(mrow[e] - mnew);
            mrow[e] = mnew;
        }
#pragma unroll
        for (int e = 0; e < 8; ++e) {
            float p0 = __expf(s0[e] - mrow[e]);
            float p1 = __expf(s1[e] - mrow[e]);
            s0[e] = p0; s1[e] = p1;
            float v = p0 + p1;
            v += __shfl_xor(v, 1, 32);
            v += __shfl_xor(v, 2, 32);
            v += __shfl_xor(v, 4, 32);
            v += __shfl_xor(v, 8, 32);
            lrow[e] = lrow[e] * alpha[e] + v;
        }
#pragma unroll
        for (int e = 0; e < 8; ++e) {
            o0[e] *= alpha[e]; o1[e] *= alpha[e];
            o2[e] *= alpha[e]; o3[e] *= alpha[e];
        }

        // ---- transpose P (C layout) -> A layout via per-wave LDS tile.
        // DS ops from one wave are in-order, no barrier needed.
#pragma unroll
        for (int e = 0; e < 8; ++e) {
            pl[(kh + e) * 32 + m]      = s0[e];
            pl[(kh + e) * 32 + 16 + m] = s1[e];
        }
        union { unsigned short s[16]; v16bf v; } pf;
#pragma unroll
        for (int e = 0; e < 8; ++e) {
            pf.s[e]     = f2bf(pl[m * 32 + kh + e]);
            pf.s[e + 8] = f2bf(pl[m * 32 + kh + 16 + e]);
        }

        // ---- PV: V^T stored [64, T] so B-fragments are contiguous loads
        v16bf vb;
        vb = load_frag(Vp, 0,  SEQ_T, k0); o0 = bwmma(pf.v, vb, o0);
        vb = load_frag(Vp, 16, SEQ_T, k0); o1 = bwmma(pf.v, vb, o1);
        vb = load_frag(Vp, 32, SEQ_T, k0); o2 = bwmma(pf.v, vb, o2);
        vb = load_frag(Vp, 48, SEQ_T, k0); o3 = bwmma(pf.v, vb, o3);
    }

    // ---- normalize and write y as [B, T, H, D] bf16 (proj GEMM input)
#pragma unroll
    for (int e = 0; e < 8; ++e) lrow[e] = 1.0f / lrow[e];
    v8f ofr[4] = { o0, o1, o2, o3 };
#pragma unroll
    for (int f = 0; f < 4; ++f) {
        int d = f * 16 + m;
#pragma unroll
        for (int e = 0; e < 8; ++e) {
            int t = q0 + kh + e;
            ybuf[(((size_t)b * SEQ_T + t) * N_HEADS + h) * HEAD_DIM + d] =
                f2bf(ofr[f][e] * lrow[e]);
        }
    }
}

// ---------------------------------------------------------------------------
// Output projection GEMM: [8192,1024]bf16 x [1024,1024] + bias -> f32 out.
// Same TDM-staged structure as the QKV GEMM.
// ---------------------------------------------------------------------------
__global__ __launch_bounds__(256) void proj_gemm_kernel(
    const unsigned short* __restrict__ yb,    // [8192,1024]
    const unsigned short* __restrict__ wT,    // [1024,1024]
    const float* __restrict__ bias,           // [1024]
    float* __restrict__ out) {                // [8192,1024]
    const int K = D_MODEL;
    __shared__ unsigned short abuf[2][32][64];

    int wave = threadIdx.x >> 5;
    int lane = threadIdx.x & 31;
    int m0 = blockIdx.x * 32;
    int n0 = blockIdx.y * 512 + wave * 64;

    v8f acc[2][4] = {};

    if (wave == 0)
        tdm_load_tile_2d(yb + (size_t)m0 * K,
                         (unsigned)(uintptr_t)(void*)&abuf[0][0][0],
                         32, 64, K, ROWS);

    for (int kc = 0; kc < K; kc += 64) {
        int buf = (kc >> 6) & 1;
        if (wave == 0) __builtin_amdgcn_s_wait_tensorcnt(0);
        __syncthreads();
        if (wave == 0 && (kc + 64) < K)
            tdm_load_tile_2d(yb + (size_t)m0 * K + kc + 64,
                             (unsigned)(uintptr_t)(void*)&abuf[buf ^ 1][0][0],
                             32, 64, K, ROWS);
        if ((kc + 64) < K)
            __builtin_prefetch(wT + (size_t)(n0 + (lane & 63)) * K + kc + 64, 0, 0);

        const unsigned short* A = &abuf[buf][0][0];
#pragma unroll
        for (int ks = 0; ks < 64; ks += 32) {
            v16bf a0 = load_frag(A, 0,  64, ks);
            v16bf a1 = load_frag(A, 16, 64, ks);
#pragma unroll
            for (int nt = 0; nt < 4; ++nt) {
                v16bf b = load_frag(wT, n0 + nt * 16, K, kc + ks);
                acc[0][nt] = bwmma(a0, b, acc[0][nt]);
                acc[1][nt] = bwmma(a1, b, acc[1][nt]);
            }
        }
    }

    int coln   = lane & 15;
    int rowoff = (lane >> 4) << 3;
#pragma unroll
    for (int mt = 0; mt < 2; ++mt) {
#pragma unroll
        for (int nt = 0; nt < 4; ++nt) {
            int ccol = n0 + nt * 16 + coln;
            float bv = bias[ccol];
#pragma unroll
            for (int e = 0; e < 8; ++e) {
                int row = m0 + mt * 16 + rowoff + e;
                out[(size_t)row * D_MODEL + ccol] = acc[mt][nt][e] + bv;
            }
        }
    }
}

// ---------------------------------------------------------------------------
// Launcher
// ---------------------------------------------------------------------------
extern "C" void kernel_launch(void* const* d_in, const int* in_sizes, int n_in,
                              void* d_out, int out_size, void* d_ws, size_t ws_size,
                              hipStream_t stream) {
    const float* x      = (const float*)d_in[0];   // [4,2048,1024]
    const float* w_qkv  = (const float*)d_in[1];   // [1024,3072]
    const float* b_qkv  = (const float*)d_in[2];   // [3072]
    const float* w_proj = (const float*)d_in[3];   // [1024,1024]
    const float* b_proj = (const float*)d_in[4];   // [1024]
    float* out = (float*)d_out;

    // workspace layout (bf16 = 2 bytes)
    char* ws = (char*)d_ws;
    unsigned short* xb     = (unsigned short*)ws; ws += (size_t)ROWS * D_MODEL * 2;      // 16 MB
    unsigned short* wqkvT  = (unsigned short*)ws; ws += (size_t)QKV_N * D_MODEL * 2;     // 6 MB
    unsigned short* wprojT = (unsigned short*)ws; ws += (size_t)D_MODEL * D_MODEL * 2;   // 2 MB
    unsigned short* qb     = (unsigned short*)ws; ws += (size_t)ROWS * D_MODEL * 2;      // 16 MB
    unsigned short* kb     = (unsigned short*)ws; ws += (size_t)ROWS * D_MODEL * 2;      // 16 MB
    unsigned short* vTb    = (unsigned short*)ws; ws += (size_t)ROWS * D_MODEL * 2;      // 16 MB
    unsigned short* yb     = (unsigned short*)ws; ws += (size_t)ROWS * D_MODEL * 2;      // 16 MB

    int nx = ROWS * D_MODEL;                      // 8388608
    f32_to_bf16_kernel<<<(nx + 255) / 256, 256, 0, stream>>>(x, xb, nx);
    int nwq = D_MODEL * QKV_N;                    // 3145728
    transpose_to_bf16_kernel<<<(nwq + 255) / 256, 256, 0, stream>>>(w_qkv, wqkvT, D_MODEL, QKV_N);
    int nwp = D_MODEL * D_MODEL;                  // 1048576
    transpose_to_bf16_kernel<<<(nwp + 255) / 256, 256, 0, stream>>>(w_proj, wprojT, D_MODEL, D_MODEL);

    // QKV projection: block = 32(M) x 512(N), 8 waves each with a 32x64 tile
    qkv_gemm_kernel<<<dim3(ROWS / 32, QKV_N / 512), 256, 0, stream>>>(
        xb, wqkvT, b_qkv, qb, kb, vTb);

    // Flash attention: 128 q-rows per block (8 waves x 16), 64 (b,h) pairs
    flash_attn_kernel<<<dim3(SEQ_T / 128, BATCH * N_HEADS), 256, 0, stream>>>(
        qb, kb, vTb, yb);

    // Output projection: block = 32(M) x 512(N)
    proj_gemm_kernel<<<dim3(ROWS / 32, D_MODEL / 512), 256, 0, stream>>>(
        yb, wprojT, b_proj, out);
}
